// GraphBlock_57844619542924
// MI455X (gfx1250) — matrix operand order repaired and checked by reading, not compiled, not run on hardware
//
#include <hip/hip_runtime.h>
#include <hip/hip_bf16.h>

// ---------------- problem constants ----------------
#define BB   64
#define TJ   243          // T (temporal nodes)
#define JJ   17           // joints
#define CC   128          // channels
#define TP   256          // T padded to 16*16 bands
#define NB   16           // number of 16-row bands
#define LDSTR 136         // bf16 row stride in LDS (pad vs bank conflicts; 272B = 17*16B)

typedef __attribute__((ext_vector_type(16))) __bf16 v16bf;
typedef __attribute__((ext_vector_type(8)))  __bf16 v8bf;
typedef __attribute__((ext_vector_type(8)))  float  v8f;

// ---------------- dynamic LDS layout (bytes) ----------------
#define OFF_DINV   0
#define OFF_ADJ    1024                              // 256 rows * 8 u32
#define OFF_YB     (OFF_ADJ  + 256*8*4)              // 9216
#define OFF_WV     (OFF_YB   + TP*LDSTR*2)           // y bf16 (243 rows + zero pad)
#define OFF_WU     (OFF_WV   + CC*LDSTR*2)           // Vw bf16
#define OFF_VO     (OFF_WU   + CC*LDSTR*2)           // Uw bf16
#define SMEM_BYTES (OFF_VO   + 244*LDSTR*2)          // V=yVw^T+b output bf16
// = 214,848 B  (< 320 KB WGP LDS)

// A-fragment (16x32 bf16): lane<16 -> row row0+lane, K {0..7,16..23}+32ks
//                          lane>=16 -> row row0+lane-16, K {8..15,24..31}+32ks
__device__ __forceinline__ v16bf frag_a(const __bf16* base, int row0, int ks, int lane) {
  const int r  = row0 + (lane & 15);
  const int ka = ks * 32 + ((lane >> 4) << 3);
  const __bf16* p = base + r * LDSTR + ka;
  v8bf lo = *(const v8bf*)(p);
  v8bf hi = *(const v8bf*)(p + 16);
  v16bf f;
#pragma unroll
  for (int i = 0; i < 8; ++i) { f[i] = lo[i]; f[i + 8] = hi[i]; }
  return f;
}

// B-fragment (32x16 bf16): lane holds column N=(lane&15); K {0..7,16..23}+8*half (+32ks)
__device__ __forceinline__ v16bf frag_b(const __bf16* base, int row0, int ks, int lane) {
  const int r  = row0 + (lane & 15);
  const int kb = ks * 32 + ((lane >> 4) << 3);
  const __bf16* p = base + r * LDSTR + kb;
  v8bf lo = *(const v8bf*)(p);
  v8bf hi = *(const v8bf*)(p + 16);
  v16bf f;
#pragma unroll
  for (int i = 0; i < 8; ++i) { f[i] = lo[i]; f[i + 8] = hi[i]; }
  return f;
}

__device__ __forceinline__ v8f wmma_bf16(v16bf a, v16bf b, v8f c) {
  return __builtin_amdgcn_wmma_f32_16x16x32_bf16(false, a, false, b, (short)0, c, false, false);
}

__global__ __launch_bounds__(256, 1)
void graphblock_kernel(const float* __restrict__ x,   const float* __restrict__ attn,
                       const float* __restrict__ lnw, const float* __restrict__ lnb,
                       const float* __restrict__ Uw,  const float* __restrict__ Ub,
                       const float* __restrict__ Vw,  const float* __restrict__ Vb,
                       const float* __restrict__ bnw, const float* __restrict__ bnb,
                       const float* __restrict__ ls1,
                       float* __restrict__ out, float* __restrict__ gfeat)
{
  extern __shared__ unsigned char smem[];
  float*    dinv = (float*)   (smem + OFF_DINV);
  unsigned* adjm = (unsigned*)(smem + OFF_ADJ);
  __bf16*   yb   = (__bf16*)  (smem + OFF_YB);
  __bf16*   wv   = (__bf16*)  (smem + OFF_WV);
  __bf16*   wu   = (__bf16*)  (smem + OFF_WU);
  __bf16*   vo   = (__bf16*)  (smem + OFF_VO);

  const int tid  = threadIdx.x;
  const int lane = tid & 31;
  const int wave = tid >> 5;
  const int half = lane >> 4;
  const int ln   = lane & 15;
  const int n = blockIdx.x;
  const int b = n / JJ;
  const int j = n % JJ;

  // ---------- Phase A: weights -> LDS bf16, LayerNorm -> yb bf16 ----------
  for (int i = tid; i < CC * CC; i += 256) {
    const int r = i >> 7, c = i & 127;
    wv[r * LDSTR + c] = (__bf16)Vw[i];
    wu[r * LDSTR + c] = (__bf16)Uw[i];
  }
  for (int i = tid; i < (TP - TJ) * LDSTR; i += 256)
    yb[TJ * LDSTR + i] = (__bf16)0.f;
  {
    const int c0 = lane * 4;
    const float4 lw = *(const float4*)(lnw + c0);
    const float4 lb = *(const float4*)(lnb + c0);
    for (int t = wave; t < TJ; t += 8) {
      const float4 v = *(const float4*)(x + (((size_t)b * TJ + t) * JJ + j) * CC + c0);
      float s  = v.x + v.y + v.z + v.w;
      float sq = v.x * v.x + v.y * v.y + v.z * v.z + v.w * v.w;
#pragma unroll
      for (int off = 16; off >= 1; off >>= 1) {
        s  += __shfl_xor(s,  off, 32);
        sq += __shfl_xor(sq, off, 32);
      }
      const float mean = s * (1.f / 128.f);
      const float var  = sq * (1.f / 128.f) - mean * mean;
      const float rs   = rsqrtf(var + 1e-5f);
      __bf16* yr = yb + t * LDSTR + c0;
      yr[0] = (__bf16)((v.x - mean) * rs * lw.x + lb.x);
      yr[1] = (__bf16)((v.y - mean) * rs * lw.y + lb.y);
      yr[2] = (__bf16)((v.z - mean) * rs * lw.z + lb.z);
      yr[3] = (__bf16)((v.w - mean) * rs * lw.w + lb.w);
    }
  }
  __syncthreads();

  // ---------- Phase B: sim band via WMMA, top-4 threshold, adjacency, dinv ----------
  for (int band = wave; band < NB; band += 8) {
    v8f acc[16] = {};
#pragma unroll
    for (int ks = 0; ks < 4; ++ks) {
      const v16bf A = frag_a(yb, band * 16, ks, lane);
#pragma unroll
      for (int st = 0; st < 16; ++st) {
        const v16bf Bf = frag_b(yb, st * 16, ks, lane);
        acc[st] = wmma_bf16(A, Bf, acc[st]);
      }
    }
    if (ln >= 3) {             // mask padded columns s = 240+ln >= 243 in tile 15
#pragma unroll
      for (int c = 0; c < 8; ++c) acc[15][c] = -__builtin_inff();
    }
    // 4th-largest per row (iterative distinct-max; ties measure-zero for random data)
    v8f cur;
#pragma unroll
    for (int c = 0; c < 8; ++c) cur[c] = __builtin_inff();
#pragma unroll
    for (int it = 0; it < 4; ++it) {
      v8f m;
#pragma unroll
      for (int c = 0; c < 8; ++c) m[c] = -__builtin_inff();
#pragma unroll
      for (int st = 0; st < 16; ++st)
#pragma unroll
        for (int c = 0; c < 8; ++c) {
          const float v = acc[st][c];
          if (v < cur[c]) m[c] = fmaxf(m[c], v);
        }
#pragma unroll
      for (int off = 8; off >= 1; off >>= 1)
#pragma unroll
        for (int c = 0; c < 8; ++c) m[c] = fmaxf(m[c], __shfl_xor(m[c], off, 16));
      cur = m;
    }
    // adjacency bitmasks + row degrees (lanes 0-15 = row band*16+c, 16-31 = +8)
#pragma unroll
    for (int c = 0; c < 8; ++c) {
      unsigned wA[8], wB[8];
#pragma unroll
      for (int w = 0; w < 8; ++w) { wA[w] = 0u; wB[w] = 0u; }
#pragma unroll
      for (int st = 0; st < 16; ++st) {
        const unsigned bal = (unsigned)__ballot(acc[st][c] >= cur[c]);
        wA[st >> 1] |= (bal & 0xFFFFu) << ((st & 1) * 16);
        wB[st >> 1] |= (bal >> 16)     << ((st & 1) * 16);
      }
      if (lane == 0) {
        const int rA = band * 16 + c, rB = rA + 8;
        int dA = 0, dB = 0;
#pragma unroll
        for (int w = 0; w < 8; ++w) {
          adjm[rA * 8 + w] = wA[w]; dA += __popc(wA[w]);
          adjm[rB * 8 + w] = wB[w]; dB += __popc(wB[w]);
        }
        if (rA < TJ) dinv[rA] = rsqrtf((float)dA);
        if (rB < TJ) dinv[rB] = rsqrtf((float)dB);
      }
    }
  }

  // ---------- Phase C: V = y @ Vw^T + Vb  -> vo (bf16) ----------
  float vbias[8];
#pragma unroll
  for (int ct = 0; ct < 8; ++ct) vbias[ct] = Vb[ct * 16 + ln];
  for (int band = wave; band < NB; band += 8) {
    v8f vacc[8] = {};
#pragma unroll
    for (int ks = 0; ks < 4; ++ks) {
      const v16bf A = frag_a(yb, band * 16, ks, lane);
#pragma unroll
      for (int ct = 0; ct < 8; ++ct) {
        const v16bf Bf = frag_b(wv, ct * 16, ks, lane);
        vacc[ct] = wmma_bf16(A, Bf, vacc[ct]);
      }
    }
#pragma unroll
    for (int c = 0; c < 8; ++c) {
      const int r = band * 16 + c + half * 8;
      if (r < TJ) {
#pragma unroll
        for (int ct = 0; ct < 8; ++ct)
          vo[r * LDSTR + ct * 16 + ln] = (__bf16)(vacc[ct][c] + vbias[ct]);
      }
    }
  }
  __syncthreads();

  // ---------- Phase D: U GEMM + sparse aggregation + fused epilogue ----------
  float ubv[8], lsv[8];
#pragma unroll
  for (int ct = 0; ct < 8; ++ct) {
    ubv[ct] = Ub[ct * 16 + ln];
    lsv[ct] = ls1[ct * 16 + ln];
  }
  const float bnrs = rsqrtf(1.f + 1e-5f);
  for (int band = wave; band < NB; band += 8) {
    v8f uacc[8] = {};
#pragma unroll
    for (int ks = 0; ks < 4; ++ks) {
      const v16bf A = frag_a(yb, band * 16, ks, lane);
#pragma unroll
      for (int ct = 0; ct < 8; ++ct) {
        const v16bf Bf = frag_b(wu, ct * 16, ks, lane);
        uacc[ct] = wmma_bf16(A, Bf, uacc[ct]);
      }
    }
#pragma unroll
    for (int c = 0; c < 8; ++c) {
      const int r = band * 16 + c + half * 8;
      if (r >= TJ) continue;
      const float di = dinv[r];
      float aggv[8];
#pragma unroll
      for (int ct = 0; ct < 8; ++ct) aggv[ct] = 0.f;
      for (int w = 0; w < 8; ++w) {
        unsigned bits = adjm[r * 8 + w];
        while (bits) {
          const int s = (w << 5) + __builtin_ctz(bits);
          bits &= bits - 1;
          const float ws = dinv[s];
          const __bf16* vp = vo + s * LDSTR;
#pragma unroll
          for (int ct = 0; ct < 8; ++ct)
            aggv[ct] += ws * (float)vp[ct * 16 + ln];
        }
      }
      const float bs  = bnw[r] * bnrs;
      const float bsh = bnb[r];
      const size_t base = (((size_t)b * TJ + r) * JJ + j) * CC;
#pragma unroll
      for (int ct = 0; ct < 8; ++ct) {
        const int col = ct * 16 + ln;
        const float yv = (float)yb[r * LDSTR + col];
        const float h  = (uacc[ct][c] + ubv[ct] + di * aggv[ct]) * bs + bsh;
        const float z  = fmaxf(yv + h, 0.f);
        const float gx = lsv[ct] * z;
        out  [base + col] = gx + attn[base + col] + x[base + col];
        gfeat[base + col] = gx * 0.5f;
      }
    }
  }
}

extern "C" void kernel_launch(void* const* d_in, const int* in_sizes, int n_in,
                              void* d_out, int out_size, void* d_ws, size_t ws_size,
                              hipStream_t stream) {
  const float* x    = (const float*)d_in[0];
  const float* attn = (const float*)d_in[1];
  const float* lnw  = (const float*)d_in[2];
  const float* lnb  = (const float*)d_in[3];
  const float* Uwp  = (const float*)d_in[4];
  const float* Ubp  = (const float*)d_in[5];
  const float* Vwp  = (const float*)d_in[6];
  const float* Vbp  = (const float*)d_in[7];
  const float* bnw  = (const float*)d_in[8];
  const float* bnb  = (const float*)d_in[9];
  const float* ls1  = (const float*)d_in[10];
  float* out   = (float*)d_out;
  float* gfeat = out + (size_t)BB * TJ * JJ * CC;

  static_assert(SMEM_BYTES <= 320 * 1024, "LDS budget");
  (void)hipFuncSetAttribute((const void*)graphblock_kernel,
                            hipFuncAttributeMaxDynamicSharedMemorySize, SMEM_BYTES);
  graphblock_kernel<<<dim3(BB * JJ), dim3(256), SMEM_BYTES, stream>>>(
      x, attn, lnw, lnb, Uwp, Ubp, Vwp, Vbp, bnw, bnb, ls1, out, gfeat);
}
